// QuantizedLinear_29386166239516
// MI455X (gfx1250) — matrix operand order repaired and checked by reading, not compiled, not run on hardware
//
#include <hip/hip_runtime.h>

// ---------------------------------------------------------------------------
// QuantizedLinear on MI455X (gfx1250):
//   out = norms[o] * ((X @ R^T) @ Yhat^T)[m,o] + bias[o],  Yhat = centroids[idx]
// Both GEMMs are TN (A row-major over K, B row-major over K) and run on
// v_wmma_f32_16x16x32_f16. LDS is double-buffered; tile k+1 is fetched
// (async-to-LDS when available, else via registers) while tile k is computed.
// ---------------------------------------------------------------------------

typedef _Float16 f16;
typedef __attribute__((ext_vector_type(16))) _Float16 v16h;
typedef __attribute__((ext_vector_type(8)))  float    v8f;
typedef int qlv4i __attribute__((vector_size(16)));   // builtin's element type

#define DIM_M   4096      // B*S
#define DIM_K   2048      // IN
#define DIM_N   8192      // OUT
#define NCODE   16

static constexpr int LDSS = 40;   // f16 stride per 32-wide LDS row (pad -> conflict-free b128 gathers)

union FragU { uint4 u[2]; v16h v; };   // 32 bytes: one WMMA A/B fragment per lane

#if defined(__has_builtin)
#  if __has_builtin(__builtin_amdgcn_global_load_async_to_lds_b128)
#    define QL_HAS_ASYNC 1
#  endif
#endif
#ifndef QL_HAS_ASYNC
#  define QL_HAS_ASYNC 0
#endif

__device__ __forceinline__ void ql_wait_async0() {
#if defined(__has_builtin) && __has_builtin(__builtin_amdgcn_s_wait_asynccnt)
  __builtin_amdgcn_s_wait_asynccnt(0);
#else
  asm volatile("s_wait_asynccnt 0" ::: "memory");
#endif
}

// --------------------------- kernel 1: dequantize ---------------------------
__global__ __launch_bounds__(256)
void QL_dequant_kernel(const int* __restrict__ idx, const float* __restrict__ centroids,
                       f16* __restrict__ yq, int total4) {
  __shared__ f16 lut[NCODE];
  if (threadIdx.x < NCODE) lut[threadIdx.x] = (f16)centroids[threadIdx.x];
  __syncthreads();
  int t = blockIdx.x * 256 + threadIdx.x;
  int stride = gridDim.x * 256;
  for (int i = t; i < total4; i += stride) {
    int4 v = reinterpret_cast<const int4*>(idx)[i];
    f16 o[4];
    o[0] = lut[v.x & 15]; o[1] = lut[v.y & 15];
    o[2] = lut[v.z & 15]; o[3] = lut[v.w & 15];
    reinterpret_cast<uint2*>(yq)[i] = *reinterpret_cast<const uint2*>(o);
  }
}

// ----------------------- staging helpers (128x32 tile) ----------------------
struct RegsF32 { float4 v[4]; };   // 256 thr * 4 * float4  = 128 rows * 32 f32
struct RegsF16 { uint4  v[2]; };   // 256 thr * 2 * 8xf16   = 128 rows * 32 f16

__device__ __forceinline__ void ql_load_f32(const float* __restrict__ src, int ld,
                                            int rowBase, int k0, int tid, RegsF32& r) {
#pragma unroll
  for (int it = 0; it < 4; ++it) {          // independent loads -> full MLP
    int i = tid + it * 256, rr = i >> 3, c = (i & 7) * 4;
    r.v[it] = *reinterpret_cast<const float4*>(src + (size_t)(rowBase + rr) * ld + k0 + c);
  }
}
__device__ __forceinline__ void ql_store_f32(const RegsF32& r, f16* __restrict__ lds, int tid) {
#pragma unroll
  for (int it = 0; it < 4; ++it) {
    int i = tid + it * 256, rr = i >> 3, c = (i & 7) * 4;
    f16* d = lds + rr * LDSS + c;           // packs to v_cvt_pk_f16_f32 + ds_store_b64
    d[0] = (f16)r.v[it].x; d[1] = (f16)r.v[it].y;
    d[2] = (f16)r.v[it].z; d[3] = (f16)r.v[it].w;
  }
}
__device__ __forceinline__ void ql_load_f16(const f16* __restrict__ src, int ld,
                                            int rowBase, int k0, int tid, RegsF16& r) {
#pragma unroll
  for (int it = 0; it < 2; ++it) {
    int i = tid + it * 256, rr = i >> 2, c = (i & 3) * 8;
    r.v[it] = *reinterpret_cast<const uint4*>(src + (size_t)(rowBase + rr) * ld + k0 + c);
  }
}
__device__ __forceinline__ void ql_store_f16(const RegsF16& r, f16* __restrict__ lds, int tid) {
#pragma unroll
  for (int it = 0; it < 2; ++it) {
    int i = tid + it * 256, rr = i >> 2, c = (i & 3) * 8;
    *reinterpret_cast<uint4*>(lds + rr * LDSS + c) = r.v[it];
  }
}

#if QL_HAS_ASYNC
// gfx1250 async copy: global -> LDS directly, tracked by ASYNCcnt.
// Builtin signature (from hipcc diagnostic): arg0 = __device__ (AS1) int4* src,
// arg1 = __shared__ (AS3) int4* dst, imm offset, imm cpol.
typedef __attribute__((address_space(1))) qlv4i* ql_gptr_t;
typedef __attribute__((address_space(3))) qlv4i* ql_lptr_t;

__device__ __forceinline__ void ql_async_f16(const f16* __restrict__ src, int ld,
                                             int rowBase, int k0, int tid, f16* lds) {
#pragma unroll
  for (int it = 0; it < 2; ++it) {
    int i = tid + it * 256, rr = i >> 2, c = (i & 3) * 8;
    const f16* g = src + (size_t)(rowBase + rr) * ld + k0 + c;
    f16* l = lds + rr * LDSS + c;
    // Generic LDS pointer's low 32 bits are the LDS byte offset (ISA flat rules).
    __builtin_amdgcn_global_load_async_to_lds_b128(
        (ql_gptr_t)(uintptr_t)g,
        (ql_lptr_t)(uint32_t)(uintptr_t)l,
        0, 0);
  }
}
#endif

// --------------------- per-wave 32x64 WMMA step (K=32) ----------------------
__device__ __forceinline__ void ql_mma_step(const f16* __restrict__ lA, const f16* __restrict__ lB,
                                            int waveM, int waveN, int hi, int lr,
                                            v8f (&acc)[2][4]) {
  FragU fa[2], fb[4];
  // A frag per ISA 16-bit 16x32 layout: lanes 0-15 K{0..7,16..23}, lanes 16-31 K{8..15,24..31}
#pragma unroll
  for (int mi = 0; mi < 2; ++mi) {
    const f16* p = &lA[(waveM * 32 + mi * 16 + lr) * LDSS + hi * 8];
    fa[mi].u[0] = *reinterpret_cast<const uint4*>(p);
    fa[mi].u[1] = *reinterpret_cast<const uint4*>(p + 16);
  }
  // B frag: lane holds column N = lr, same K order as A: lanes 0-15 K{0..7,16..23}, lanes 16-31 K{8..15,24..31}
#pragma unroll
  for (int ni = 0; ni < 4; ++ni) {
    const f16* p = &lB[(waveN * 64 + ni * 16 + lr) * LDSS + hi * 8];
    fb[ni].u[0] = *reinterpret_cast<const uint4*>(p);
    fb[ni].u[1] = *reinterpret_cast<const uint4*>(p + 16);
  }
#pragma unroll
  for (int mi = 0; mi < 2; ++mi)
#pragma unroll
    for (int ni = 0; ni < 4; ++ni)
      acc[mi][ni] = __builtin_amdgcn_wmma_f32_16x16x32_f16(
          false, fa[mi].v, false, fb[ni].v, (short)0, acc[mi][ni], false, false);
}

// ------------------------- GEMM: D = A @ B^T (TN) --------------------------
// Block tile 128x128, 8 wave32s as 4(M) x 2(N), each wave 32x64 (2x4 WMMA tiles).
template <typename TIN, bool FINAL>
__global__ __launch_bounds__(256)
void QL_gemm_tn_kernel(const TIN* __restrict__ A, const TIN* __restrict__ Bm,
                       void* __restrict__ Out,
                       const float* __restrict__ scale, const float* __restrict__ bias,
                       int M, int N, int K) {
  __shared__ __align__(16) f16 lA[2][128 * LDSS];   // 2 x 10 KB
  __shared__ __align__(16) f16 lB[2][128 * LDSS];   // 2 x 10 KB

  const int tid   = threadIdx.x;
  const int lane  = tid & 31;
  const int wv    = tid >> 5;
  const int waveM = wv & 3;
  const int waveN = wv >> 2;
  const int hi    = lane >> 4;
  const int lr    = lane & 15;
  const int blockN0 = blockIdx.x * 128;
  const int blockM0 = blockIdx.y * 128;

  v8f acc[2][4];
#pragma unroll
  for (int mi = 0; mi < 2; ++mi)
#pragma unroll
    for (int ni = 0; ni < 4; ++ni) acc[mi][ni] = (v8f)0.0f;

  const int nk = K >> 5;

#if QL_HAS_ASYNC
  if constexpr (sizeof(TIN) == 2) {
    // Fully-async pipeline: tile k+1 streams into the other LDS buffer while
    // the WMMAs consume tile k; one barrier per K-step.
    ql_async_f16((const f16*)A,  K, blockM0, 0, tid, lA[0]);
    ql_async_f16((const f16*)Bm, K, blockN0, 0, tid, lB[0]);
    ql_wait_async0();
    __syncthreads();
    for (int kt = 0; kt < nk; ++kt) {
      const int cur = kt & 1;
      if (kt + 1 < nk) {
        ql_async_f16((const f16*)A,  K, blockM0, (kt + 1) << 5, tid, lA[cur ^ 1]);
        ql_async_f16((const f16*)Bm, K, blockN0, (kt + 1) << 5, tid, lB[cur ^ 1]);
      }
      ql_mma_step(lA[cur], lB[cur], waveM, waveN, hi, lr, acc);
      if (kt + 1 < nk) ql_wait_async0();
      __syncthreads();
    }
  } else
#endif
  if constexpr (sizeof(TIN) == 4) {
    RegsF32 ra, rb;
    ql_load_f32((const float*)A,  K, blockM0, 0, tid, ra);
    ql_load_f32((const float*)Bm, K, blockN0, 0, tid, rb);
    ql_store_f32(ra, lA[0], tid);
    ql_store_f32(rb, lB[0], tid);
    __syncthreads();
    for (int kt = 0; kt < nk; ++kt) {
      const int cur = kt & 1;
      if (kt + 1 < nk) {                       // issue k+1 loads before compute
        ql_load_f32((const float*)A,  K, blockM0, (kt + 1) << 5, tid, ra);
        ql_load_f32((const float*)Bm, K, blockN0, (kt + 1) << 5, tid, rb);
      }
      ql_mma_step(lA[cur], lB[cur], waveM, waveN, hi, lr, acc);
      if (kt + 1 < nk) {                       // convert+store after compute
        ql_store_f32(ra, lA[cur ^ 1], tid);
        ql_store_f32(rb, lB[cur ^ 1], tid);
      }
      __syncthreads();
    }
  } else {
    RegsF16 ra, rb;
    ql_load_f16((const f16*)A,  K, blockM0, 0, tid, ra);
    ql_load_f16((const f16*)Bm, K, blockN0, 0, tid, rb);
    ql_store_f16(ra, lA[0], tid);
    ql_store_f16(rb, lB[0], tid);
    __syncthreads();
    for (int kt = 0; kt < nk; ++kt) {
      const int cur = kt & 1;
      if (kt + 1 < nk) {
        ql_load_f16((const f16*)A,  K, blockM0, (kt + 1) << 5, tid, ra);
        ql_load_f16((const f16*)Bm, K, blockN0, (kt + 1) << 5, tid, rb);
      }
      ql_mma_step(lA[cur], lB[cur], waveM, waveN, hi, lr, acc);
      if (kt + 1 < nk) {
        ql_store_f16(ra, lA[cur ^ 1], tid);
        ql_store_f16(rb, lB[cur ^ 1], tid);
      }
      __syncthreads();
    }
  }

  // Epilogue. C/D layout: VGPR r -> M = r (lanes 0-15) / 8+r (lanes 16-31), N = lane&15.
#pragma unroll
  for (int mi = 0; mi < 2; ++mi) {
#pragma unroll
    for (int ni = 0; ni < 4; ++ni) {
      const int col  = blockN0 + waveN * 64 + ni * 16 + lr;
      const int row0 = blockM0 + waveM * 32 + mi * 16 + hi * 8;
      if constexpr (FINAL) {
        const float sc = scale[col];
        const float bi = bias[col];
        float* out = reinterpret_cast<float*>(Out);
#pragma unroll
        for (int r = 0; r < 8; ++r)
          out[(size_t)(row0 + r) * N + col] = acc[mi][ni][r] * sc + bi;
      } else {
        f16* out = reinterpret_cast<f16*>(Out);
#pragma unroll
        for (int r = 0; r < 8; ++r)
          out[(size_t)(row0 + r) * N + col] = (f16)acc[mi][ni][r];
      }
    }
  }
}

// --------------------------------- launch ----------------------------------
extern "C" void kernel_launch(void* const* d_in, const int* in_sizes, int n_in,
                              void* d_out, int out_size, void* d_ws, size_t ws_size,
                              hipStream_t stream) {
  (void)in_sizes; (void)n_in; (void)out_size; (void)ws_size;
  const float* x         = (const float*)d_in[0];   // [4096, 2048]
  const int*   widx      = (const int*)  d_in[1];   // [8192, 2048]
  const float* norms     = (const float*)d_in[2];   // [8192]
  const float* centroids = (const float*)d_in[3];   // [16]
  const float* rot       = (const float*)d_in[4];   // [2048, 2048]
  const float* bias      = (const float*)d_in[5];   // [8192]

  f16* yq = (f16*)d_ws;                             // [8192, 2048] f16  (32 MB)
  f16* zb = yq + (size_t)DIM_N * DIM_K;             // [4096, 2048] f16  (16 MB)

  // 1) dequantize indices -> f16 Yhat
  QL_dequant_kernel<<<4096, 256, 0, stream>>>(widx, centroids, yq, DIM_N * DIM_K / 4);

  // 2) Z = X @ R^T   (f32 inputs staged as f16, f16 output)
  dim3 g1(DIM_K / 128, DIM_M / 128);
  QL_gemm_tn_kernel<float, false><<<g1, 256, 0, stream>>>(
      x, rot, zb, nullptr, nullptr, DIM_M, DIM_K, DIM_K);

  // 3) out = (Z @ Yhat^T) * norms + bias   (f32 output, fused epilogue)
  dim3 g2(DIM_N / 128, DIM_M / 128);
  QL_gemm_tn_kernel<f16, true><<<g2, 256, 0, stream>>>(
      zb, yq, d_out, norms, bias, DIM_M, DIM_N, DIM_K);
}
